// GATv2Model_21620865368799
// MI455X (gfx1250) — matrix operand order repaired and checked by reading, not compiled, not run on hardware
//
#include <hip/hip_runtime.h>
#include <math.h>

// ---------------------------------------------------------------------------
// GATv2 + edge classifier for MI455X (gfx1250), wave32, WMMA bf16.
// Dense projections: v_wmma_f32_16x16x32_bf16 with b128 fragment loads
// (A chunked from row-major, B pre-swizzled into fragment order).
// Irregular graph ops: float atomics. Aw1/Aw2 branch skipped (softmax over a
// size-1 axis == 1.0 identically).
// ---------------------------------------------------------------------------

typedef __bf16 bf16_t;
typedef __bf16 v16bf __attribute__((ext_vector_type(16)));
typedef __bf16 v8bf  __attribute__((ext_vector_type(8)));
typedef float  v8f   __attribute__((ext_vector_type(8)));

__device__ __forceinline__ float elu1f(float v) {
  return v > 0.f ? v : (__expf(v) - 1.f);
}

__device__ __forceinline__ void atomicMaxF(float* addr, float val) {
  if (val >= 0.f) atomicMax((int*)addr, __float_as_int(val));
  else            atomicMin((unsigned int*)addr, __float_as_uint(val));
}

__device__ __forceinline__ v16bf cat8(v8bf lo, v8bf hi) {
  return __builtin_shufflevector(lo, hi, 0, 1, 2, 3, 4, 5, 6, 7,
                                 8, 9, 10, 11, 12, 13, 14, 15);
}

// ---------------- utility kernels ----------------
__global__ __launch_bounds__(256) void k_fill_f32(float* __restrict__ p, float v, int n) {
  int i = blockIdx.x * blockDim.x + threadIdx.x;
  if (i < n) p[i] = v;
}

__global__ __launch_bounds__(256) void k_f32_to_bf16(const float* __restrict__ s,
                                                     bf16_t* __restrict__ d, int n) {
  int i = blockIdx.x * blockDim.x + threadIdx.x;
  if (i < n) d[i] = (bf16_t)s[i];
}

// Pack a K x N (row-major, f32) weight matrix into WMMA B-fragment order:
// Bp[ct][kk][lane][e] with col = ct*16 + (lane&15), k = kk*32 + (lane<16?0:8) + (e<8 ? e : e+8).
// Zero-pads k >= K. Each lane's 16 bf16 are then contiguous (2 x b128 loads).
__global__ __launch_bounds__(256) void k_pack_b(const float* __restrict__ B,
                                                bf16_t* __restrict__ Bp,
                                                int N, int K, int ksteps) {
  int i = blockIdx.x * blockDim.x + threadIdx.x;
  int total = (N >> 4) * ksteps * 512;
  if (i >= total) return;
  int e    = i & 15;
  int lane = (i >> 4) & 31;
  int rest = i >> 9;
  int kk   = rest % ksteps;
  int ct   = rest / ksteps;
  int col  = ct * 16 + (lane & 15);
  int k    = kk * 32 + ((lane < 16) ? 0 : 8) + ((e < 8) ? e : e + 8);
  Bp[i] = (k < K) ? (bf16_t)B[k * N + col] : (bf16_t)0.f;
}

// ---------------- self-loop attr (PyG fill_value='mean') ----------------
__global__ __launch_bounds__(256) void k_deg_loopsum(const float* __restrict__ ea,
                                                     const int* __restrict__ dst,
                                                     float* __restrict__ deg,
                                                     float* __restrict__ loopsum, int E) {
  int e = blockIdx.x * blockDim.x + threadIdx.x;
  if (e >= E) return;
  int d = dst[e];
  atomicAdd(&deg[d], 1.f);
#pragma unroll
  for (int k = 0; k < 16; k++)
    atomicAdd(&loopsum[d * 16 + k], ea[e * 16 + k]);
}

__global__ __launch_bounds__(256) void k_loop_div(float* __restrict__ loopsum,
                                                  const float* __restrict__ deg, int Nn) {
  int i = blockIdx.x * blockDim.x + threadIdx.x;
  if (i < Nn * 16) loopsum[i] /= fmaxf(deg[i >> 4], 1.f);
}

__global__ __launch_bounds__(256) void k_build_easl(const float* __restrict__ ea,
                                                    const float* __restrict__ loopat,
                                                    bf16_t* __restrict__ o, int E, int Nn) {
  int i = blockIdx.x * blockDim.x + threadIdx.x;
  int tot = (E + Nn) * 16;
  if (i >= tot) return;
  int e = i >> 4, k = i & 15;
  float v = (e < E) ? ea[i] : loopat[(e - E) * 16 + k];
  o[i] = (bf16_t)v;
}

// ---------------- bf16 WMMA GEMM: C[M,N] = A[M,K] @ B[K,N] + bias ----------
// K is a template parameter (K % 16 == 0) so chunk guards fold at compile
// time. A: row-major bf16; Bp: fragment-packed (see k_pack_b).
// One 16x16 tile per wave, 4 waves/block cover 64 columns.
template <int K>
__global__ __launch_bounds__(128) void k_gemm_bf16(const bf16_t* __restrict__ A,
                                                   const bf16_t* __restrict__ Bp,
                                                   const float* __restrict__ bias,
                                                   float* __restrict__ C, int M, int N) {
  constexpr int KSTEPS = (K + 31) / 32;
  int wave = threadIdx.x >> 5;
  int lane = threadIdx.x & 31;
  int row0 = blockIdx.x * 16;
  int ct   = blockIdx.y * 4 + wave;
  int col0 = ct * 16;
  if (row0 >= M || col0 >= N) return;

  int arow = row0 + (lane & 15);
  int col  = col0 + (lane & 15);
  int kbA  = (lane < 16) ? 0 : 8;

  const bf16_t* Abase = A + (size_t)arow * K;
  const v8bf*   Bv    = (const v8bf*)Bp;
  size_t        bbase = ((size_t)ct * KSTEPS * 32 + lane) * 2;  // v8bf units

  v8f acc = {};
  v8bf z8 = {};
#pragma unroll
  for (int kk = 0; kk < KSTEPS; kk++) {
    int ka0 = kk * 32 + kbA;
    v8bf alo = (kk * 32 + 16 <= K) ? *(const v8bf*)(Abase + ka0)      : z8;
    v8bf ahi = (kk * 32 + 32 <= K) ? *(const v8bf*)(Abase + ka0 + 16) : z8;
    v8bf blo = Bv[bbase + (size_t)kk * 64];
    v8bf bhi = Bv[bbase + (size_t)kk * 64 + 1];
    acc = __builtin_amdgcn_wmma_f32_16x16x32_bf16(false, cat8(alo, ahi),
                                                  false, cat8(blo, bhi),
                                                  (short)0, acc, false, false);
  }

  float bv = bias ? bias[col] : 0.f;
#pragma unroll
  for (int v = 0; v < 8; v++) {
    int r = row0 + v + ((lane < 16) ? 0 : 8);
    C[(size_t)r * N + col] = acc[v] + bv;
  }
}

// ---------------- GATv2 attention (per layer) ----------------
__global__ __launch_bounds__(256) void k_edge_logits(const float* __restrict__ xl,
                                                     const float* __restrict__ xr,
                                                     const float* __restrict__ eproj,
                                                     const int* __restrict__ src,
                                                     const int* __restrict__ dst,
                                                     const float* __restrict__ att,
                                                     float* __restrict__ alog,
                                                     float* __restrict__ amax,
                                                     int E, int Nn) {
  int e = blockIdx.x * blockDim.x + threadIdx.x;
  if (e >= E + Nn) return;
  int s, d;
  if (e < E) { s = src[e]; d = dst[e]; } else { s = d = e - E; }
  const float4* pl = (const float4*)(xl + (size_t)s * 64);
  const float4* pr = (const float4*)(xr + (size_t)d * 64);
  const float4* pe = (const float4*)(eproj + (size_t)e * 64);
  const float4* pa = (const float4*)att;
  float acc = 0.f;
#pragma unroll
  for (int c = 0; c < 16; c++) {
    float4 l = pl[c], r = pr[c], ep = pe[c], at = pa[c];
    float m;
    m = l.x + r.x + ep.x; m = m > 0.f ? m : 0.2f * m; acc += m * at.x;
    m = l.y + r.y + ep.y; m = m > 0.f ? m : 0.2f * m; acc += m * at.y;
    m = l.z + r.z + ep.z; m = m > 0.f ? m : 0.2f * m; acc += m * at.z;
    m = l.w + r.w + ep.w; m = m > 0.f ? m : 0.2f * m; acc += m * at.w;
  }
  alog[e] = acc;
  atomicMaxF(&amax[d], acc);
}

__global__ __launch_bounds__(256) void k_exp_den(const float* __restrict__ alog,
                                                 const int* __restrict__ dst,
                                                 const float* __restrict__ amax,
                                                 float* __restrict__ exv,
                                                 float* __restrict__ den, int E, int Nn) {
  int e = blockIdx.x * blockDim.x + threadIdx.x;
  if (e >= E + Nn) return;
  int d = (e < E) ? dst[e] : e - E;
  float x = __expf(alog[e] - amax[d]);
  exv[e] = x;
  atomicAdd(&den[d], x);
}

__global__ __launch_bounds__(256) void k_scatter(const float* __restrict__ xl,
                                                 const float* __restrict__ exv,
                                                 const float* __restrict__ den,
                                                 const int* __restrict__ src,
                                                 const int* __restrict__ dst,
                                                 float* __restrict__ acc, int E, int Nn) {
  long i = (long)blockIdx.x * blockDim.x + threadIdx.x;
  if (i >= (long)(E + Nn) * 64) return;
  int e = (int)(i >> 6), c = (int)(i & 63);
  int s, d;
  if (e < E) { s = src[e]; d = dst[e]; } else { s = d = e - E; }
  float alpha = exv[e] / den[d];
  atomicAdd(&acc[d * 64 + c], xl[(size_t)s * 64 + c] * alpha);
}

__global__ __launch_bounds__(256) void k_finalize(float* __restrict__ h,
                                                  const float* __restrict__ bias,
                                                  bf16_t* __restrict__ hb, int Nn, int act) {
  int i = blockIdx.x * blockDim.x + threadIdx.x;
  if (i >= Nn * 64) return;
  float v = h[i] + bias[i & 63];
  if (act == 1) v = elu1f(v);
  h[i] = v;
  hb[i] = (bf16_t)v;
}

// ---------------- fused edge classifier ----------------
// out[e] = elu(ef @ Cw1 + Cb1) @ Cw2 + Cb2,  ef = [h2[src]|h2[dst]|edge_attr]
// 512 threads = 16 waves per block, 16 edges per block. ef tile staged in LDS
// (stride 160 for 16B-aligned chunk loads); Cw1 is fragment-packed (Kp=160).
__global__ __launch_bounds__(512) void k_classifier(const bf16_t* __restrict__ h2b,
                                                    const float* __restrict__ ea,
                                                    const int* __restrict__ src,
                                                    const int* __restrict__ dst,
                                                    const bf16_t* __restrict__ Cw1p,
                                                    const float* __restrict__ Cb1,
                                                    const float* __restrict__ Cw2,
                                                    const float* __restrict__ Cb2,
                                                    float* __restrict__ out, int E) {
  __shared__ __align__(16) bf16_t efs[16][160];
  __shared__ float sout[16];
  const int tid = threadIdx.x;
  const int e0 = blockIdx.x * 16;

  for (int idx = tid; idx < 16 * 144; idx += 512) {
    int r = idx / 144, k = idx - r * 144;
    int e = e0 + r;
    float v = 0.f;
    if (e < E) {
      if (k < 64)       v = (float)h2b[(size_t)src[e] * 64 + k];
      else if (k < 128) v = (float)h2b[(size_t)dst[e] * 64 + (k - 64)];
      else              v = ea[(size_t)e * 16 + (k - 128)];
    }
    efs[r][k] = (bf16_t)v;
  }
  if (tid < 16) sout[tid] = 0.f;
  __syncthreads();

  const int wave = tid >> 5, lane = tid & 31;  // wave = column tile (16 x 16 of 256)
  const int col  = wave * 16 + (lane & 15);
  const int kbA  = (lane < 16) ? 0 : 8;
  const bf16_t* Abase = &efs[lane & 15][0];
  const v8bf*   Bv    = (const v8bf*)Cw1p;
  size_t        bbase = ((size_t)wave * 5 * 32 + lane) * 2;  // KSTEPS = 5

  v8f acc = {};
  v8bf z8 = {};
#pragma unroll
  for (int kk = 0; kk < 5; kk++) {
    int ka0 = kk * 32 + kbA;
    v8bf alo = (kk * 32 + 16 <= 144) ? *(const v8bf*)(Abase + ka0)      : z8;
    v8bf ahi = (kk * 32 + 32 <= 144) ? *(const v8bf*)(Abase + ka0 + 16) : z8;
    v8bf blo = Bv[bbase + (size_t)kk * 64];
    v8bf bhi = Bv[bbase + (size_t)kk * 64 + 1];
    acc = __builtin_amdgcn_wmma_f32_16x16x32_bf16(false, cat8(alo, ahi),
                                                  false, cat8(blo, bhi),
                                                  (short)0, acc, false, false);
  }

  float cb = Cb1[col], cw = Cw2[col];
  float p[8];
#pragma unroll
  for (int v = 0; v < 8; v++) p[v] = elu1f(acc[v] + cb) * cw;

#pragma unroll
  for (int m = 8; m >= 1; m >>= 1)
#pragma unroll
    for (int v = 0; v < 8; v++) p[v] += __shfl_xor(p[v], m, 32);

  if ((lane & 15) == 0) {  // lane 0 -> rows 0..7, lane 16 -> rows 8..15
    int rb = (lane < 16) ? 0 : 8;
#pragma unroll
    for (int v = 0; v < 8; v++) atomicAdd(&sout[rb + v], p[v]);
  }
  __syncthreads();
  if (tid < 16 && e0 + tid < E) out[e0 + tid] = sout[tid] + Cb2[0];
}

// ---------------------------------------------------------------------------
static inline int cdiv(long a, long b) { return (int)((a + b - 1) / b); }

extern "C" void kernel_launch(void* const* d_in, const int* in_sizes, int n_in,
                              void* d_out, int out_size, void* d_ws, size_t ws_size,
                              hipStream_t stream) {
  (void)n_in; (void)out_size; (void)ws_size;
  const float* x         = (const float*)d_in[0];
  const float* edge_attr = (const float*)d_in[1];
  const float* W1l = (const float*)d_in[2];  const float* b1l   = (const float*)d_in[3];
  const float* W1r = (const float*)d_in[4];  const float* b1r   = (const float*)d_in[5];
  const float* We1 = (const float*)d_in[6];  const float* att1  = (const float*)d_in[7];
  const float* bias1 = (const float*)d_in[8];
  const float* W2l = (const float*)d_in[9];  const float* b2l   = (const float*)d_in[10];
  const float* W2r = (const float*)d_in[11]; const float* b2r   = (const float*)d_in[12];
  const float* We2 = (const float*)d_in[13]; const float* att2  = (const float*)d_in[14];
  const float* bias2 = (const float*)d_in[15];
  // d_in[16..19] = Aw1/Ab1/Aw2/Ab2: softmax over a size-1 axis == 1.0 -> unused
  const float* Cw1 = (const float*)d_in[20]; const float* Cb1   = (const float*)d_in[21];
  const float* Cw2 = (const float*)d_in[22]; const float* Cb2   = (const float*)d_in[23];
  const int*   eidx = (const int*)d_in[24];

  const int Nn  = in_sizes[0] / 128;   // 50000
  const int E   = in_sizes[24] / 2;    // 800000
  const int Esl = E + Nn;
  const int* src = eidx;
  const int* dst = eidx + E;
  float* out = (float*)d_out;

  // ---- workspace bump allocator (256B aligned) ----
  char* wsb = (char*)d_ws;
  size_t off = 0;
  auto alloc = [&](size_t bytes) -> void* {
    void* p = wsb + off;
    off += (bytes + 255) & ~(size_t)255;
    return p;
  };
  float*  deg    = (float*) alloc((size_t)Nn * 4);
  float*  loopat = (float*) alloc((size_t)Nn * 16 * 4);
  bf16_t* easlb  = (bf16_t*)alloc((size_t)Esl * 16 * 2);
  bf16_t* xb     = (bf16_t*)alloc((size_t)Nn * 128 * 2);
  float*  xl     = (float*) alloc((size_t)Nn * 64 * 4);
  float*  xr     = (float*) alloc((size_t)Nn * 64 * 4);
  float*  eproj  = (float*) alloc((size_t)Esl * 64 * 4);
  float*  alog   = (float*) alloc((size_t)Esl * 4);
  float*  exv    = (float*) alloc((size_t)Esl * 4);
  float*  amax   = (float*) alloc((size_t)Nn * 4);
  float*  den    = (float*) alloc((size_t)Nn * 4);
  float*  h      = (float*) alloc((size_t)Nn * 64 * 4);
  bf16_t* hb     = (bf16_t*)alloc((size_t)Nn * 64 * 2);
  float*  h2     = (float*) alloc((size_t)Nn * 64 * 4);
  bf16_t* h2b    = (bf16_t*)alloc((size_t)Nn * 64 * 2);
  // fragment-packed weights: (N/16)*ksteps*512 bf16 each
  bf16_t* W1l_p  = (bf16_t*)alloc(4 * 4 * 512 * 2);   // N=64 K=128
  bf16_t* W1r_p  = (bf16_t*)alloc(4 * 4 * 512 * 2);
  bf16_t* We1_p  = (bf16_t*)alloc(4 * 1 * 512 * 2);   // N=64 K=16
  bf16_t* W2l_p  = (bf16_t*)alloc(4 * 2 * 512 * 2);   // N=64 K=64
  bf16_t* W2r_p  = (bf16_t*)alloc(4 * 2 * 512 * 2);
  bf16_t* We2_p  = (bf16_t*)alloc(4 * 1 * 512 * 2);
  bf16_t* Cw1_p  = (bf16_t*)alloc(16 * 5 * 512 * 2);  // N=256 K=144

  const int T = 256;
  // ---- conversions / weight packing ----
  k_f32_to_bf16<<<cdiv((long)Nn * 128, T), T, 0, stream>>>(x, xb, Nn * 128);
  k_pack_b<<<cdiv(4 * 4 * 512, T), T, 0, stream>>>(W1l, W1l_p, 64, 128, 4);
  k_pack_b<<<cdiv(4 * 4 * 512, T), T, 0, stream>>>(W1r, W1r_p, 64, 128, 4);
  k_pack_b<<<cdiv(4 * 1 * 512, T), T, 0, stream>>>(We1, We1_p, 64, 16, 1);
  k_pack_b<<<cdiv(4 * 2 * 512, T), T, 0, stream>>>(W2l, W2l_p, 64, 64, 2);
  k_pack_b<<<cdiv(4 * 2 * 512, T), T, 0, stream>>>(W2r, W2r_p, 64, 64, 2);
  k_pack_b<<<cdiv(4 * 1 * 512, T), T, 0, stream>>>(We2, We2_p, 64, 16, 1);
  k_pack_b<<<cdiv(16 * 5 * 512, T), T, 0, stream>>>(Cw1, Cw1_p, 256, 144, 5);

  // ---- self-loop edge attrs ----
  k_fill_f32<<<cdiv(Nn, T), T, 0, stream>>>(deg, 0.f, Nn);
  k_fill_f32<<<cdiv((long)Nn * 16, T), T, 0, stream>>>(loopat, 0.f, Nn * 16);
  k_deg_loopsum<<<cdiv(E, T), T, 0, stream>>>(edge_attr, dst, deg, loopat, E);
  k_loop_div<<<cdiv((long)Nn * 16, T), T, 0, stream>>>(loopat, deg, Nn);
  k_build_easl<<<cdiv((long)Esl * 16, T), T, 0, stream>>>(edge_attr, loopat, easlb, E, Nn);

  // ---- GAT layer 1 ----
  dim3 gB(128);
  k_gemm_bf16<128><<<dim3(cdiv(Nn, 16), 1),  gB, 0, stream>>>(xb, W1l_p, b1l, xl, Nn, 64);
  k_gemm_bf16<128><<<dim3(cdiv(Nn, 16), 1),  gB, 0, stream>>>(xb, W1r_p, b1r, xr, Nn, 64);
  k_gemm_bf16<16><<<dim3(cdiv(Esl, 16), 1), gB, 0, stream>>>(easlb, We1_p, nullptr, eproj, Esl, 64);

  k_fill_f32<<<cdiv(Nn, T), T, 0, stream>>>(amax, -INFINITY, Nn);
  k_fill_f32<<<cdiv(Nn, T), T, 0, stream>>>(den, 0.f, Nn);
  k_fill_f32<<<cdiv((long)Nn * 64, T), T, 0, stream>>>(h, 0.f, Nn * 64);
  k_edge_logits<<<cdiv(Esl, T), T, 0, stream>>>(xl, xr, eproj, src, dst, att1, alog, amax, E, Nn);
  k_exp_den<<<cdiv(Esl, T), T, 0, stream>>>(alog, dst, amax, exv, den, E, Nn);
  k_scatter<<<cdiv((long)Esl * 64, T), T, 0, stream>>>(xl, exv, den, src, dst, h, E, Nn);
  k_finalize<<<cdiv((long)Nn * 64, T), T, 0, stream>>>(h, bias1, hb, Nn, 1);  // +bias1, ELU

  // ---- GAT layer 2 ----
  k_gemm_bf16<64><<<dim3(cdiv(Nn, 16), 1),  gB, 0, stream>>>(hb, W2l_p, b2l, xl, Nn, 64);
  k_gemm_bf16<64><<<dim3(cdiv(Nn, 16), 1),  gB, 0, stream>>>(hb, W2r_p, b2r, xr, Nn, 64);
  k_gemm_bf16<16><<<dim3(cdiv(Esl, 16), 1), gB, 0, stream>>>(easlb, We2_p, nullptr, eproj, Esl, 64);

  k_fill_f32<<<cdiv(Nn, T), T, 0, stream>>>(amax, -INFINITY, Nn);
  k_fill_f32<<<cdiv(Nn, T), T, 0, stream>>>(den, 0.f, Nn);
  k_fill_f32<<<cdiv((long)Nn * 64, T), T, 0, stream>>>(h2, 0.f, Nn * 64);
  k_edge_logits<<<cdiv(Esl, T), T, 0, stream>>>(xl, xr, eproj, src, dst, att2, alog, amax, E, Nn);
  k_exp_den<<<cdiv(Esl, T), T, 0, stream>>>(alog, dst, amax, exv, den, E, Nn);
  k_scatter<<<cdiv((long)Esl * 64, T), T, 0, stream>>>(xl, exv, den, src, dst, h2, E, Nn);
  k_finalize<<<cdiv((long)Nn * 64, T), T, 0, stream>>>(h2, bias2, h2b, Nn, 0);  // +bias2 only

  // ---- fused edge classifier ----
  k_classifier<<<cdiv(E, 16), 512, 0, stream>>>(h2b, edge_attr, src, dst,
                                                Cw1_p, Cb1, Cw2, Cb2, out, E);
}
